// EdgePredictor_11776800325801
// MI455X (gfx1250) — matrix operand order repaired and checked by reading, not compiled, not run on hardware
//
#include <hip/hip_runtime.h>

typedef __attribute__((ext_vector_type(16))) __bf16 v16bf;
typedef __attribute__((ext_vector_type(8)))  __bf16 v8bf;
typedef __attribute__((ext_vector_type(8)))  float  v8f;
typedef __attribute__((ext_vector_type(4)))  float  v4f;

#define D_IN   256
#define D_HID  128
#define PITCH  264      // bf16 elems per LDS row: 256 + 8 pad -> 528B row (4 banks mod 64)
#define M_TILE 64
#define NWAVES 4

// Layer-1 GEMM with gathered rows, bf16 WMMA, f32 accumulate; layer-2 + softmax in regs.
__global__ __launch_bounds__(128) void edge_mlp_kernel(
    const float* __restrict__ src_h, const float* __restrict__ dst_h,
    const int*   __restrict__ src_idx, const int* __restrict__ dst_idx,
    const float* __restrict__ W1, const float* __restrict__ b1,
    const float* __restrict__ W2, const float* __restrict__ b2,
    float* __restrict__ out, int E)
{
    __shared__ alignas(16) __bf16 lds_w1t[D_HID * PITCH];     // W1^T [n][k], 67584 B
    __shared__ alignas(16) __bf16 lds_x[NWAVES][16 * PITCH];  // per-wave X [m][k], 33792 B

    const int tid  = threadIdx.x;
    const int lane = tid & 31;
    const int wv   = tid >> 5;
    const int j    = lane & 15;   // column / row-in-half index
    const int half = lane >> 4;   // 0 or 1

    // ---- Stage W1 (f32 -> bf16, transpose into [n][k]) -----------------
    {
        const int n = tid;                       // 128 threads == 128 columns
        for (int k = 0; k < D_IN; k += 2) {      // coalesced global reads per iter
            float a = W1[(size_t)k       * D_HID + n];
            float c = W1[(size_t)(k + 1) * D_HID + n];
            lds_w1t[n * PITCH + k]     = (__bf16)a;
            lds_w1t[n * PITCH + k + 1] = (__bf16)c;
        }
    }

    // ---- Stage X: gather this wave's 16 edges (src||dst), f32 -> bf16 --
    const long long edgeBase = (long long)blockIdx.x * M_TILE + wv * 16;
    const float* tbl = half ? dst_h : src_h;     // lanes 0-15: src half, 16-31: dst half
    const int*   idx = half ? dst_idx : src_idx;
    for (int m = 0; m < 16; ++m) {
        long long e = edgeBase + m;
        if (e >= E) e = E - 1;                   // clamp (tail-safe gather)
        int node = idx[e];
        const float* p = tbl + (size_t)node * 128 + j * 8;
        v4f f0 = *(const v4f*)p;
        v4f f1 = *(const v4f*)(p + 4);
        v8bf bv;
        bv[0]=(__bf16)f0[0]; bv[1]=(__bf16)f0[1]; bv[2]=(__bf16)f0[2]; bv[3]=(__bf16)f0[3];
        bv[4]=(__bf16)f1[0]; bv[5]=(__bf16)f1[1]; bv[6]=(__bf16)f1[2]; bv[7]=(__bf16)f1[3];
        *(v8bf*)&lds_x[wv][m * PITCH + half * 128 + j * 8] = bv;
    }
    __syncthreads();

    // ---- Per-lane constants: b1 (acc init) and W2 columns --------------
    float bn[8], w2a[8], w2b[8];
    #pragma unroll
    for (int nt = 0; nt < 8; ++nt) {
        int n  = nt * 16 + j;
        bn[nt]  = b1[n];
        w2a[nt] = W2[n * 2 + 0];
        w2b[nt] = W2[n * 2 + 1];
    }

    v8f acc[8];
    #pragma unroll
    for (int nt = 0; nt < 8; ++nt) {
        v8f c; 
        #pragma unroll
        for (int i = 0; i < 8; ++i) c[i] = bn[nt];   // same N for all 8 rows of a lane
        acc[nt] = c;
    }

    // ---- Layer 1: 16x128 = sum_k A(16x32) x B(32x16), 64 WMMA ----------
    const __bf16* xrow = &lds_x[wv][0];
    #pragma unroll
    for (int kc = 0; kc < 8; ++kc) {
        // A frag: row m=j; lanes 0-15 K=0..15, lanes 16-31 K=16..31
        const __bf16* pa = xrow + j * PITCH + kc * 32 + half * 16;
        v8bf a0 = *(const v8bf*)pa;
        v8bf a1 = *(const v8bf*)(pa + 8);
        v16bf A;
        #pragma unroll
        for (int i = 0; i < 8; ++i) { A[i] = a0[i]; A[i + 8] = a1[i]; }

        #pragma unroll
        for (int nt = 0; nt < 8; ++nt) {
            // B frag: col n=nt*16+j; lanes 0-15 K=0..15, lanes 16-31 K=16..31
            const __bf16* pb = lds_w1t + (nt * 16 + j) * PITCH + kc * 32 + half * 16;
            v8bf b0 = *(const v8bf*)pb;
            v8bf b1v = *(const v8bf*)(pb + 8);
            v16bf B;
            #pragma unroll
            for (int i = 0; i < 8; ++i) { B[i] = b0[i]; B[i + 8] = b1v[i]; }
            acc[nt] = __builtin_amdgcn_wmma_f32_16x16x32_bf16(
                false, A, false, B, (short)0, acc[nt], false, false);
        }
    }

    // ---- Layer 2 (relu -> 128x2 dot) from accumulators ------------------
    // acc[nt][v] = h[row v + 8*half][col nt*16 + j]
    float p0[8], p1[8];
    #pragma unroll
    for (int v = 0; v < 8; ++v) { p0[v] = 0.f; p1[v] = 0.f; }
    #pragma unroll
    for (int nt = 0; nt < 8; ++nt) {
        #pragma unroll
        for (int v = 0; v < 8; ++v) {
            float h = acc[nt][v];
            h = h > 0.f ? h : 0.f;               // relu
            p0[v] = fmaf(h, w2a[nt], p0[v]);
            p1[v] = fmaf(h, w2b[nt], p1[v]);
        }
    }
    // reduce over the 16 lanes of each half-wave (xor masks stay in-half)
    #pragma unroll
    for (int off = 1; off < 16; off <<= 1) {
        #pragma unroll
        for (int v = 0; v < 8; ++v) {
            p0[v] += __shfl_xor(p0[v], off, 32);
            p1[v] += __shfl_xor(p1[v], off, 32);
        }
    }

    // ---- softmax(2) + store: lane j -> row v=j&7, output o=j>>3 --------
    {
        int v = j & 7;
        int o = j >> 3;
        long long e = edgeBase + v + 8 * half;
        if (e < (long long)E) {
            float s0 = p0[v] + b2[0];
            float s1 = p1[v] + b2[1];
            float mx = fmaxf(s0, s1);
            float e0 = __expf(s0 - mx);
            float e1 = __expf(s1 - mx);
            float r  = (o == 0 ? e0 : e1) / (e0 + e1);
            out[e * 2 + o] = r;
        }
    }
}

extern "C" void kernel_launch(void* const* d_in, const int* in_sizes, int n_in,
                              void* d_out, int out_size, void* d_ws, size_t ws_size,
                              hipStream_t stream) {
    const float* src_h  = (const float*)d_in[0];
    const float* dst_h  = (const float*)d_in[1];
    const int*   src_idx = (const int*)d_in[2];
    const int*   dst_idx = (const int*)d_in[3];
    const float* W1 = (const float*)d_in[4];
    const float* b1 = (const float*)d_in[5];
    const float* W2 = (const float*)d_in[6];
    const float* b2 = (const float*)d_in[7];
    float* out = (float*)d_out;

    int E = in_sizes[2];                         // edge count
    int grid = (E + M_TILE - 1) / M_TILE;
    edge_mlp_kernel<<<grid, 128, 0, stream>>>(src_h, dst_h, src_idx, dst_idx,
                                              W1, b1, W2, b2, out, E);
}
